// Seq2SeqEncDec_63969242907223
// MI455X (gfx1250) — matrix operand order repaired and checked by reading, not compiled, not run on hardware
//
#include <hip/hip_runtime.h>

#define H 1024
#define T_STEPS 512
#define P_STEPS 96
#define B_SZ 256
#define NBLK 64

typedef __attribute__((ext_vector_type(16))) _Float16 v16h;
typedef __attribute__((ext_vector_type(8)))  _Float16 v8h;
typedef __attribute__((ext_vector_type(8)))  float    v8f;

union V16 { v16h v; v8h h[2]; };

__device__ __forceinline__ float fsigm(float x) {
  return 1.0f / (1.0f + __expf(-x));
}
__device__ __forceinline__ float ftanh(float x) {
  float e = __expf(-2.0f * x);
  return (1.0f - e) / (1.0f + e);
}

// Device-wide sense/generation barrier. 64 blocks of 256 threads are
// guaranteed co-resident on MI455X, so spin-wait is safe.
__device__ __forceinline__ void grid_sync(unsigned* cnt, unsigned* gen) {
  __syncthreads();
  if (threadIdx.x == 0) {
    __threadfence();
    unsigned g = __hip_atomic_load(gen, __ATOMIC_RELAXED, __HIP_MEMORY_SCOPE_AGENT);
    unsigned arr = __hip_atomic_fetch_add(cnt, 1u, __ATOMIC_ACQ_REL, __HIP_MEMORY_SCOPE_AGENT);
    if (arr == NBLK - 1u) {
      __hip_atomic_store(cnt, 0u, __ATOMIC_RELAXED, __HIP_MEMORY_SCOPE_AGENT);
      __hip_atomic_fetch_add(gen, 1u, __ATOMIC_RELEASE, __HIP_MEMORY_SCOPE_AGENT);
    } else {
      while (__hip_atomic_load(gen, __ATOMIC_ACQUIRE, __HIP_MEMORY_SCOPE_AGENT) == g) {
        __builtin_amdgcn_s_sleep(2);
      }
    }
    __threadfence();
  }
  __syncthreads();
}

// One LSTM timestep for this WG's 64-row x 64-hidden-col tile.
// gates = h@Whh^T (WMMA f16->f32) + x*Wih + b, then fused cell update.
// K-loop is software-pipelined with register double buffering; unrolling
// is disabled so operands stay in registers (no scratch spills).
template <bool DO_PROJ>
__device__ __forceinline__ void lstm_step(
    const _Float16* __restrict__ hcur, _Float16* __restrict__ hnxt,
    const _Float16* __restrict__ W16, float* __restrict__ cbuf,
    const float* xs, const float* wih_s, const float* bias_s, float* hstage,
    int b0, int j0, int wr, int wc, int hi, int ln)
{
  v8f acc[4][2];
  const v8f vzero = {0.f, 0.f, 0.f, 0.f, 0.f, 0.f, 0.f, 0.f};
#pragma unroll
  for (int G = 0; G < 4; ++G)
#pragma unroll
    for (int t = 0; t < 2; ++t) acc[G][t] = vzero;

  // A fragment: lane ln = row, hi selects K sub-block (per 16-bit A 16x32 layout)
  const _Float16* aptr = hcur + (size_t)(b0 + wr * 16 + ln) * H + hi * 8;
  // B fragments: one base pointer; gate stride G*H*H (2MB) and tile stride
  // 16*H (32KB) both fit the 24-bit signed instruction offset.
  const _Float16* bptr = W16 + (size_t)(j0 + wc * 32 + ln) * H + hi * 8;

  V16 aR[2];
  V16 bR[2][8];

  auto loadA = [&](int buf, int k0) __attribute__((always_inline)) {
    aR[buf].h[0] = *(const v8h*)(aptr + k0);        // K = k0+hi*8 .. +7
    aR[buf].h[1] = *(const v8h*)(aptr + k0 + 16);   // K = k0+16+hi*8 .. +7
  };
  auto loadB = [&](int buf, int k0) __attribute__((always_inline)) {
#pragma unroll
    for (int G = 0; G < 4; ++G) {
#pragma unroll
      for (int t = 0; t < 2; ++t) {
        const _Float16* p = bptr + (size_t)G * (H * H) + t * (16 * H) + k0;
        bR[buf][G * 2 + t].h[0] = *(const v8h*)(p);
        bR[buf][G * 2 + t].h[1] = *(const v8h*)(p + 16);
      }
    }
  };
  auto mma = [&](int buf) __attribute__((always_inline)) {
#pragma unroll
    for (int G = 0; G < 4; ++G) {
#pragma unroll
      for (int t = 0; t < 2; ++t) {
        acc[G][t] = __builtin_amdgcn_wmma_f32_16x16x32_f16(
            false, aR[buf].v, false, bR[buf][G * 2 + t].v,
            (short)0, acc[G][t], false, false);
      }
    }
  };

  loadA(0, 0);
  loadB(0, 0);
#pragma unroll 1
  for (int k0 = 0; k0 < H; k0 += 64) {
    loadA(1, k0 + 32);          // prefetch slice k0+32 while computing k0
    loadB(1, k0 + 32);
    mma(0);
    if (k0 + 64 < H) {          // prefetch slice k0+64 while computing k0+32
      loadA(0, k0 + 64);
      loadB(0, k0 + 64);
    }
    mma(1);
  }

  // Cell update: all 4 gate values for (row,col) live in the same lane/slot.
#pragma unroll
  for (int t = 0; t < 2; ++t) {
    int col_l = wc * 32 + t * 16 + ln;
#pragma unroll
    for (int e = 0; e < 8; ++e) {
      int row_l = wr * 16 + hi * 8 + e;
      float xv = xs[row_l];
      float gi = acc[0][t][e] + xv * wih_s[      col_l] + bias_s[      col_l];
      float gf = acc[1][t][e] + xv * wih_s[ 64 + col_l] + bias_s[ 64 + col_l];
      float gg = acc[2][t][e] + xv * wih_s[128 + col_l] + bias_s[128 + col_l];
      float go = acc[3][t][e] + xv * wih_s[192 + col_l] + bias_s[192 + col_l];
      float iv = fsigm(gi), fv = fsigm(gf), gv = ftanh(gg), ov = fsigm(go);
      size_t idx = (size_t)(b0 + row_l) * H + (size_t)(j0 + col_l);
      float cn = fv * cbuf[idx] + iv * gv;
      float hn = ov * ftanh(cn);
      cbuf[idx] = cn;
      hnxt[idx] = (_Float16)hn;
      if (DO_PROJ) hstage[row_l * 64 + col_l] = hn;
    }
  }
}

__global__ __launch_bounds__(256, 1) void Seq2SeqEncDec_persistent(
    const float* __restrict__ input_seq,
    const float* __restrict__ enc_Wih, const float* __restrict__ enc_b,
    const float* __restrict__ dec_Wih, const float* __restrict__ dec_b,
    const float* __restrict__ out_W,  const float* __restrict__ out_b,
    const _Float16* __restrict__ encW16, const _Float16* __restrict__ decW16,
    _Float16* __restrict__ h16a, _Float16* __restrict__ h16b,
    float* __restrict__ cbuf, float* __restrict__ partial,
    unsigned* __restrict__ syncbuf, float* __restrict__ out)
{
  __shared__ float xs[64];
  __shared__ float wih_s[256];
  __shared__ float bias_s[256];
  __shared__ float hstage[64 * 64];

  const int tid = threadIdx.x;
  const int wg  = blockIdx.x;      // 0..63
  const int bi  = wg & 3;          // row block (4 x 64 rows)
  const int ji  = wg >> 2;         // hidden-col block (16 x 64 cols)
  const int b0  = bi * 64;
  const int j0  = ji * 64;
  const int lane = tid & 31;
  const int wave = tid >> 5;       // 8 waves
  const int wr = wave & 3;         // 4 row groups of 16
  const int wc = wave >> 2;        // 2 col groups of 32
  const int hi = lane >> 4;        // half-wave
  const int ln = lane & 15;
  unsigned* cnt = syncbuf;
  unsigned* gen = syncbuf + 1;
  const float ob = out_b[0];

  // ---- encoder phase ----
  {
    int G = tid >> 6, cl = tid & 63;
    wih_s[tid]  = enc_Wih[G * H + j0 + cl];
    bias_s[tid] = enc_b [G * H + j0 + cl];
  }
  __syncthreads();

  for (int t = 0; t < T_STEPS; ++t) {
    const _Float16* hc = (t & 1) ? h16b : h16a;
    _Float16*       hn = (t & 1) ? h16a : h16b;
    if (tid < 64) xs[tid] = input_seq[(size_t)(b0 + tid) * T_STEPS + t];
    __syncthreads();
    lstm_step<false>(hc, hn, encW16, cbuf, xs, wih_s, bias_s, hstage,
                     b0, j0, wr, wc, hi, ln);
    grid_sync(cnt, gen);
  }

  // ---- decoder phase ----
  {
    int G = tid >> 6, cl = tid & 63;
    wih_s[tid]  = dec_Wih[G * H + j0 + cl];
    bias_s[tid] = dec_b [G * H + j0 + cl];
  }
  __syncthreads();

  for (int p = 0; p < P_STEPS; ++p) {
    int s = T_STEPS + p;
    const _Float16* hc = (s & 1) ? h16b : h16a;
    _Float16*       hn = (s & 1) ? h16a : h16b;
    if (tid < 64) {
      float dv;
      if (p == 0) {
        dv = input_seq[(size_t)(b0 + tid) * T_STEPS + (T_STEPS - 1)];
      } else {
        // deterministic fixed-order reduction of 16 per-WG partials
        const float* pp = partial + (size_t)((p - 1) & 1) * (B_SZ * 16)
                        + (size_t)(b0 + tid) * 16;
        float a = 0.f;
#pragma unroll
        for (int q = 0; q < 16; ++q) a += pp[q];
        dv = a + ob;
        if (ji == 0) out[(size_t)(b0 + tid) * P_STEPS + (p - 1)] = dv;
      }
      xs[tid] = dv;
    }
    __syncthreads();
    lstm_step<true>(hc, hn, decW16, cbuf, xs, wih_s, bias_s, hstage,
                    b0, j0, wr, wc, hi, ln);
    __syncthreads();
    if (tid < 64) {  // partial projection over this WG's 64 hidden cols
      float a = 0.f;
      const float* ow = out_W + j0;
      for (int c2 = 0; c2 < 64; ++c2) a += hstage[tid * 64 + c2] * ow[c2];
      partial[(size_t)(p & 1) * (B_SZ * 16) + (size_t)(b0 + tid) * 16 + ji] = a;
    }
    grid_sync(cnt, gen);
  }

  // final decoder output (p = P-1)
  if (ji == 0 && tid < 64) {
    const float* pp = partial + (size_t)((P_STEPS - 1) & 1) * (B_SZ * 16)
                    + (size_t)(b0 + tid) * 16;
    float a = 0.f;
#pragma unroll
    for (int q = 0; q < 16; ++q) a += pp[q];
    out[(size_t)(b0 + tid) * P_STEPS + (P_STEPS - 1)] = a + ob;
  }
}

// Per-call init: weights -> f16, h/c state from h0/c0, barrier reset.
__global__ void seq2seq_init(const float* __restrict__ encWhh,
                             const float* __restrict__ decWhh,
                             const float* __restrict__ h0,
                             const float* __restrict__ c0,
                             _Float16* __restrict__ encW16,
                             _Float16* __restrict__ decW16,
                             _Float16* __restrict__ h16a,
                             float* __restrict__ cbuf,
                             unsigned* __restrict__ syncbuf) {
  size_t i = (size_t)blockIdx.x * blockDim.x + threadIdx.x;
  const size_t NW = (size_t)4 * H * H;
  if (i < NW) {
    encW16[i] = (_Float16)encWhh[i];
    decW16[i] = (_Float16)decWhh[i];
  }
  if (i < (size_t)B_SZ * H) {
    h16a[i] = (_Float16)h0[i & (H - 1)];
    cbuf[i] = c0[i & (H - 1)];
  }
  if (i < 2) syncbuf[i] = 0u;
}

extern "C" void kernel_launch(void* const* d_in, const int* in_sizes, int n_in,
                              void* d_out, int out_size, void* d_ws, size_t ws_size,
                              hipStream_t stream) {
  (void)in_sizes; (void)n_in; (void)out_size; (void)ws_size;
  const float* input_seq = (const float*)d_in[0];
  const float* enc_Wih   = (const float*)d_in[1];
  const float* enc_Whh   = (const float*)d_in[2];
  const float* enc_b     = (const float*)d_in[3];
  const float* h0        = (const float*)d_in[4];
  const float* c0        = (const float*)d_in[5];
  const float* dec_Wih   = (const float*)d_in[6];
  const float* dec_Whh   = (const float*)d_in[7];
  const float* dec_b     = (const float*)d_in[8];
  const float* out_W     = (const float*)d_in[9];
  const float* out_b     = (const float*)d_in[10];

  char* ws = (char*)d_ws;
  size_t off = 0;
  unsigned* syncbuf = (unsigned*)(ws + off); off += 256;
  _Float16* encW16 = (_Float16*)(ws + off); off += (size_t)4 * H * H * sizeof(_Float16);
  _Float16* decW16 = (_Float16*)(ws + off); off += (size_t)4 * H * H * sizeof(_Float16);
  _Float16* h16a   = (_Float16*)(ws + off); off += (size_t)B_SZ * H * sizeof(_Float16);
  _Float16* h16b   = (_Float16*)(ws + off); off += (size_t)B_SZ * H * sizeof(_Float16);
  float*    cbuf   = (float*)(ws + off);    off += (size_t)B_SZ * H * sizeof(float);
  float*    partial= (float*)(ws + off);    off += (size_t)2 * B_SZ * 16 * sizeof(float);

  const size_t NW = (size_t)4 * H * H;
  seq2seq_init<<<(unsigned)((NW + 255) / 256), 256, 0, stream>>>(
      enc_Whh, dec_Whh, h0, c0, encW16, decW16, h16a, cbuf, syncbuf);

  Seq2SeqEncDec_persistent<<<NBLK, 256, 0, stream>>>(
      input_seq, enc_Wih, enc_b, dec_Wih, dec_b, out_W, out_b,
      encW16, decW16, h16a, h16b, cbuf, partial, syncbuf, (float*)d_out);
}
